// MultiHeadAttention_65532611002781
// MI455X (gfx1250) — matrix operand order repaired and checked by reading, not compiled, not run on hardware
//
#include <hip/hip_runtime.h>

// ---------------------------------------------------------------------------
// Transformer encoder layer (MHA + FC + 2x LayerNorm) for MI455X / gfx1250.
// All GEMMs on v_wmma_f32_16x16x32_bf16 (bf16 in, fp32 accumulate).
// Attention staging via GLOBAL_LOAD_ASYNC_TO_LDS (ASYNCcnt) when available,
// double-buffered in LDS so async copies overlap WMMA + softmax.
// H=8, D=512, DK=DV=64, B=4, L=2048.
// ---------------------------------------------------------------------------

typedef __attribute__((ext_vector_type(16))) __bf16 v16bf;
typedef __attribute__((ext_vector_type(8)))  float  v8f;
typedef int v4i __attribute__((vector_size(16)));

struct U32x8 { uint4 lo, hi; };

static __device__ __forceinline__ v16bf make_v16bf(uint4 lo, uint4 hi) {
  U32x8 t{lo, hi};
  return __builtin_bit_cast(v16bf, t);
}

// fp32 -> bf16 (round to nearest even)
static __device__ __forceinline__ unsigned short f2bf(float f) {
  unsigned u = __builtin_bit_cast(unsigned, f);
  unsigned r = u + 0x7FFFu + ((u >> 16) & 1u);
  return (unsigned short)(r >> 16);
}

static __device__ __forceinline__ v8f wmma_bf16(v16bf a, v16bf b, v8f c) {
  return __builtin_amdgcn_wmma_f32_16x16x32_bf16(
      /*neg_a=*/false, a, /*neg_b=*/false, b,
      /*c_mod=*/(short)0, c, /*reuse_a=*/false, /*reuse_b=*/false);
}

// A-matrix 16x32 bf16 load (contiguous k order, same as ldB):
//   lanes 0-15 : M=lane,    K = k0+0..15 (contiguous)
//   lanes16-31 : M=lane-16, K = k0+16..31 (contiguous)
static __device__ __forceinline__ v16bf ldA(const unsigned short* rowptr, int k0, int lane) {
  int kofs = k0 + ((lane >= 16) ? 16 : 0);
  uint4 lo = *(const uint4*)(rowptr + kofs);
  uint4 hi = *(const uint4*)(rowptr + kofs + 8);
  return make_v16bf(lo, hi);
}

// B-matrix 32x16 bf16 load (contiguous k order, same as ldA):
//   lanes 0-15 : N=lane,    K = k0+0..15 (contiguous)
//   lanes16-31 : N=lane-16, K = k0+16..31 (contiguous)
static __device__ __forceinline__ v16bf ldB(const unsigned short* rowptr, int k0, int lane) {
  int kofs = k0 + ((lane >= 16) ? 16 : 0);
  uint4 lo = *(const uint4*)(rowptr + kofs);
  uint4 hi = *(const uint4*)(rowptr + kofs + 8);
  return make_v16bf(lo, hi);
}

// ---------------------------------------------------------------------------
// gfx1250 async global->LDS copy (ASYNCcnt path) with sync fallback.
// Prototype (from clang diagnostic): b128 takes int-vec4 pointers, b32 int.
// ---------------------------------------------------------------------------
#if __has_builtin(__builtin_amdgcn_global_load_async_to_lds_b128)
#define HAS_ASYNC_LDS 1
#else
#define HAS_ASYNC_LDS 0
#endif

#if HAS_ASYNC_LDS
static __device__ __forceinline__ void async_cp128(const void* g, void* l) {
  __builtin_amdgcn_global_load_async_to_lds_b128(
      (__attribute__((address_space(1))) v4i*)(v4i*)(void*)g,
      (__attribute__((address_space(3))) v4i*)(v4i*)l, 0, 0);
}
static __device__ __forceinline__ void async_cp32(const void* g, void* l) {
  __builtin_amdgcn_global_load_async_to_lds_b32(
      (__attribute__((address_space(1))) int*)(int*)(void*)g,
      (__attribute__((address_space(3))) int*)(int*)l, 0, 0);
}
#endif

static __device__ __forceinline__ void wait_async0() {
#if HAS_ASYNC_LDS
#if __has_builtin(__builtin_amdgcn_s_wait_asynccnt)
  __builtin_amdgcn_s_wait_asynccnt(0);
#else
  asm volatile("s_wait_asynccnt 0x0" ::: "memory");
#endif
#endif
}

// ---------------------------------------------------------------------------
// fp32 -> bf16 elementwise conversion
// ---------------------------------------------------------------------------
__global__ void cvt_bf16_kernel(const float* __restrict__ in,
                                unsigned short* __restrict__ out, int n) {
  int i = blockIdx.x * blockDim.x + threadIdx.x;
  if (i < n) out[i] = f2bf(in[i]);
}

// ---------------------------------------------------------------------------
// GEMM: out = X[8192 x 512] * W^T  (W is [512 out][512 in], K-major rows)
// One wave computes a 16(M) x 64(N) strip: A loaded once per K-step, reused
// across 4 B tiles -> 64 WMMA / wave. 8 waves/block, grid = 512 blocks.
// mode 0: store bf16 head-major qh/kh:  out[n=h*4+b][l][dk]
// mode 1: store bf16 transposed vhT:    out[n][dk][l]
// mode 2: store fp32 plain:             out[row][col]
// ---------------------------------------------------------------------------
__global__ __launch_bounds__(256) void gemm_bf16_kernel(
    const unsigned short* __restrict__ X, const unsigned short* __restrict__ W,
    void* __restrict__ outp, int mode) {
  const int tid  = threadIdx.x;
  const int lane = tid & 31;
  const int wave = tid >> 5;
  const int gid  = blockIdx.x * 8 + wave;     // 0..4095
  const int mtile = gid >> 3;                 // 0..511 (rows of 16)
  const int ng    = gid & 7;                  // 0..7   (N-group of 64 = head)
  const int ln = lane & 15;
  const int hf = lane >> 4;

  const unsigned short* xrow = X + (size_t)(mtile * 16 + ln) * 512;
  const unsigned short* wrow0 = W + (size_t)(ng * 64 + 0 * 16 + ln) * 512;
  const unsigned short* wrow1 = W + (size_t)(ng * 64 + 1 * 16 + ln) * 512;
  const unsigned short* wrow2 = W + (size_t)(ng * 64 + 2 * 16 + ln) * 512;
  const unsigned short* wrow3 = W + (size_t)(ng * 64 + 3 * 16 + ln) * 512;

  v8f c0 = {0.f,0.f,0.f,0.f,0.f,0.f,0.f,0.f};
  v8f c1 = c0, c2 = c0, c3 = c0;
#pragma unroll
  for (int k0 = 0; k0 < 512; k0 += 32) {
    v16bf a = ldA(xrow, k0, lane);
    c0 = wmma_bf16(a, ldB(wrow0, k0, lane), c0);
    c1 = wmma_bf16(a, ldB(wrow1, k0, lane), c1);
    c2 = wmma_bf16(a, ldB(wrow2, k0, lane), c2);
    c3 = wmma_bf16(a, ldB(wrow3, k0, lane), c3);
  }

  const int row0 = mtile * 16;
  const int bb = row0 >> 11;          // batch
  const int l0 = row0 & 2047;         // seq pos base
  const int hh = ng;                  // head (N-group == head for D=512)
  v8f cc[4] = {c0, c1, c2, c3};

  if (mode == 0) {  // qh / kh : [n][l][64] bf16
    unsigned short* out = (unsigned short*)outp;
#pragma unroll
    for (int j = 0; j < 4; j++) {
      const int dk = j * 16 + ln;
      unsigned short* p =
          out + ((size_t)(hh * 4 + bb) * 2048 + l0 + hf * 8) * 64 + dk;
#pragma unroll
      for (int r = 0; r < 8; r++) p[(size_t)r * 64] = f2bf(cc[j][r]);
    }
  } else if (mode == 1) {  // vhT : [n][dk][l] bf16, 8 contiguous l per lane
    unsigned short* out = (unsigned short*)outp;
#pragma unroll
    for (int j = 0; j < 4; j++) {
      const int dk = j * 16 + ln;
      unsigned short* p =
          out + ((size_t)(hh * 4 + bb) * 64 + dk) * 2048 + (l0 + hf * 8);
      uint4 pk;
      pk.x = f2bf(cc[j][0]) | ((unsigned)f2bf(cc[j][1]) << 16);
      pk.y = f2bf(cc[j][2]) | ((unsigned)f2bf(cc[j][3]) << 16);
      pk.z = f2bf(cc[j][4]) | ((unsigned)f2bf(cc[j][5]) << 16);
      pk.w = f2bf(cc[j][6]) | ((unsigned)f2bf(cc[j][7]) << 16);
      *(uint4*)p = pk;
    }
  } else {  // fp32 plain [8192][512]
    float* out = (float*)outp;
#pragma unroll
    for (int j = 0; j < 4; j++) {
      const int nout = ng * 64 + j * 16 + ln;
      float* p = out + (size_t)(row0 + hf * 8) * 512 + nout;
#pragma unroll
      for (int r = 0; r < 8; r++) p[(size_t)r * 512] = cc[j][r];
    }
  }
}

// ---------------------------------------------------------------------------
// Flash attention. Computes S^T = K*Q^T so softmax stats are per-lane-column.
// Block: 256 thr (8 waves). Wave owns 16 queries; block covers 128 queries.
// 2048 keys in steps of 32; K-tile, V^T-tile, mask double-buffered in LDS and
// staged with async global->LDS copies overlapped with WMMA compute.
// grid = (32 headbatch, 16 query strips)
// ---------------------------------------------------------------------------
#define LOG2E 1.44269504f

__global__ __launch_bounds__(256) void attn_kernel(
    const unsigned short* __restrict__ qh, const unsigned short* __restrict__ kh,
    const unsigned short* __restrict__ vhT, const int* __restrict__ mask,
    float* __restrict__ attn_out) {
  __shared__ unsigned short kt[2][32 * 72];  // 32 keys x 64 dims (+pad)
  __shared__ unsigned short vt[2][64 * 40];  // 64 dims x 32 keys (+pad)
  __shared__ int mk[2][32];

  const int tid  = threadIdx.x;
  const int lane = tid & 31;
  const int wave = tid >> 5;
  const int ln = lane & 15;
  const int hf = lane >> 4;

  const int n = blockIdx.x;        // head*4 + batch
  const int h = n >> 2, b = n & 3;
  const int q0 = blockIdx.y * 128 + wave * 16;

  const unsigned short* kbase = kh + (size_t)n * 2048 * 64;
  const unsigned short* vbase = vhT + (size_t)n * 64 * 2048;
  const int* mbase = mask + n * 2048;

  // cooperative staging of one 32-key tile into buffer `buf`
  auto stage = [&](int k0, int buf) {
    const int r = tid >> 3, cc = tid & 7;   // K tile: 32 rows x 8 chunks
    const int rv = tid >> 2, cv = tid & 3;  // V^T tile: 64 rows x 4 chunks
#if HAS_ASYNC_LDS
    async_cp128(&kbase[(size_t)(k0 + r) * 64 + cc * 8], &kt[buf][r * 72 + cc * 8]);
    async_cp128(&vbase[(size_t)rv * 2048 + k0 + cv * 8], &vt[buf][rv * 40 + cv * 8]);
    if (tid < 32) async_cp32(&mbase[k0 + tid], &mk[buf][tid]);
#else
    *(uint4*)&kt[buf][r * 72 + cc * 8] =
        *(const uint4*)&kbase[(size_t)(k0 + r) * 64 + cc * 8];
    *(uint4*)&vt[buf][rv * 40 + cv * 8] =
        *(const uint4*)&vbase[(size_t)rv * 2048 + k0 + cv * 8];
    if (tid < 32) mk[buf][tid] = mbase[k0 + tid];
#endif
  };

  // Q^T as B-operand, kept in registers for the whole key loop.
  const unsigned short* qrow = qh + ((size_t)n * 2048 + q0 + ln) * 64;
  v16bf bq0 = ldB(qrow, 0, lane);   // dims  0..31
  v16bf bq1 = ldB(qrow, 32, lane);  // dims 32..63

  v8f o0 = {0.f,0.f,0.f,0.f,0.f,0.f,0.f,0.f};
  v8f o1 = o0, o2 = o0, o3 = o0;
  float m = -1e30f, l = 0.f;

  stage(0, 0);
  int buf = 0;

  for (int k0 = 0; k0 < 2048; k0 += 32) {
    wait_async0();     // this wave's copies into `buf` have landed
    __syncthreads();   // every wave's copies into `buf` are visible
    if (k0 + 32 < 2048) stage(k0 + 32, buf ^ 1);  // overlap with compute

    // S^T tiles: M = keys, N = queries.  4 WMMA (2 key tiles x 2 K-steps).
    v8f c0 = {0.f,0.f,0.f,0.f,0.f,0.f,0.f,0.f};
    v8f c1 = c0;
    {
      v16bf a;
      a = ldA(&kt[buf][(0 * 16 + ln) * 72], 0, lane);  c0 = wmma_bf16(a, bq0, c0);
      a = ldA(&kt[buf][(0 * 16 + ln) * 72], 32, lane); c0 = wmma_bf16(a, bq1, c0);
      a = ldA(&kt[buf][(1 * 16 + ln) * 72], 0, lane);  c1 = wmma_bf16(a, bq0, c1);
      a = ldA(&kt[buf][(1 * 16 + ln) * 72], 32, lane); c1 = wmma_bf16(a, bq1, c1);
    }

    // Online softmax (per-lane column = one query; combine halves via shfl 16)
    float s[16];
#pragma unroll
    for (int r = 0; r < 8; r++) { s[r] = c0[r]; s[8 + r] = c1[r]; }
    float bm = -1e30f;
#pragma unroll
    for (int i = 0; i < 16; i++) {
      const int key = ((i >> 3) * 16) + (i & 7) + hf * 8;
      s[i] = mk[buf][key] ? s[i] * 0.125f : -1.25e8f;  // mask, then 1/sqrt(64)
      bm = fmaxf(bm, s[i]);
    }
    bm = fmaxf(bm, __shfl_xor(bm, 16));
    const float mn = fmaxf(m, bm);
    const float alpha = exp2f((m - mn) * LOG2E);
    float ls = 0.f;
#pragma unroll
    for (int i = 0; i < 16; i++) {
      s[i] = exp2f((s[i] - mn) * LOG2E);
      ls += s[i];
    }
    ls += __shfl_xor(ls, 16);
    l = l * alpha + ls;
    m = mn;
#pragma unroll
    for (int r = 0; r < 8; r++) {
      o0[r] *= alpha; o1[r] *= alpha; o2[r] *= alpha; o3[r] *= alpha;
    }

    // P^T (C layout) -> B-operand layout: one half-wave exchange + bf16 pack.
    float e[16];
#pragma unroll
    for (int r = 0; r < 8; r++) {
      const float x0 = __shfl_xor(s[r], 16);
      const float x1 = __shfl_xor(s[8 + r], 16);
      e[r]     = hf ? x1 : s[r];
      e[8 + r] = hf ? s[8 + r] : x0;
    }
    unsigned pw[8];
#pragma unroll
    for (int j = 0; j < 8; j++)
      pw[j] = f2bf(e[2 * j]) | ((unsigned)f2bf(e[2 * j + 1]) << 16);
    uint4 plo = {pw[0], pw[1], pw[2], pw[3]};
    uint4 phi = {pw[4], pw[5], pw[6], pw[7]};
    v16bf bP = make_v16bf(plo, phi);

    // O^T += V^T * P^T : 4 WMMA (dims 0..63 in 4 tiles of 16)
    {
      v16bf a;
      a = ldA(&vt[buf][(0 * 16 + ln) * 40], 0, lane); o0 = wmma_bf16(a, bP, o0);
      a = ldA(&vt[buf][(1 * 16 + ln) * 40], 0, lane); o1 = wmma_bf16(a, bP, o1);
      a = ldA(&vt[buf][(2 * 16 + ln) * 40], 0, lane); o2 = wmma_bf16(a, bP, o2);
      a = ldA(&vt[buf][(3 * 16 + ln) * 40], 0, lane); o3 = wmma_bf16(a, bP, o3);
    }
    buf ^= 1;
  }

  const float invl = 1.f / l;
  const int q = q0 + ln;
  float* orow = attn_out + ((size_t)b * 2048 + q) * 512 + h * 64;
#pragma unroll
  for (int r = 0; r < 8; r++) {
    const int d = r + hf * 8;
    orow[d]      = o0[r] * invl;
    orow[16 + d] = o1[r] * invl;
    orow[32 + d] = o2[r] * invl;
    orow[48 + d] = o3[r] * invl;
  }
}

// ---------------------------------------------------------------------------
// LayerNorm over D=512. Wave per row; 8 rows per block (1024 blocks).
// y = (x + resid [+ bias] - mu) * rsqrt(var+eps) * g + b
// ---------------------------------------------------------------------------
__global__ __launch_bounds__(256) void ln_kernel(
    const float* __restrict__ x, const float* __restrict__ resid,
    const float* __restrict__ bias, const float* __restrict__ g,
    const float* __restrict__ bt, float* __restrict__ outF,
    unsigned short* __restrict__ outBF) {
  const int lane = threadIdx.x & 31;
  const int wave = threadIdx.x >> 5;
  const size_t row = (size_t)blockIdx.x * 8 + wave;
  const float* xr = x + row * 512;
  const float* rr = resid + row * 512;

  float t[16];
  float sum = 0.f, sq = 0.f;
#pragma unroll
  for (int i = 0; i < 16; i++) {
    const int idx = lane + i * 32;
    float v = xr[idx] + rr[idx];
    if (bias) v += bias[idx];
    t[i] = v;
    sum += v;
    sq += v * v;
  }
#pragma unroll
  for (int d = 1; d < 32; d <<= 1) {
    sum += __shfl_xor(sum, d);
    sq  += __shfl_xor(sq, d);
  }
  const float mu = sum * (1.f / 512.f);
  const float var = sq * (1.f / 512.f) - mu * mu;
  const float rs = rsqrtf(var + 1e-5f);
#pragma unroll
  for (int i = 0; i < 16; i++) {
    const int idx = lane + i * 32;
    const float y = (t[i] - mu) * rs * g[idx] + bt[idx];
    if (outF)  outF[row * 512 + idx] = y;
    if (outBF) outBF[row * 512 + idx] = f2bf(y);
  }
}

// ---------------------------------------------------------------------------
// Host-side orchestration
// ---------------------------------------------------------------------------
extern "C" void kernel_launch(void* const* d_in, const int* in_sizes, int n_in,
                              void* d_out, int out_size, void* d_ws, size_t ws_size,
                              hipStream_t stream) {
  const float* q    = (const float*)d_in[0];
  const float* k    = (const float*)d_in[1];
  const float* v    = (const float*)d_in[2];
  const int*   mask = (const int*)d_in[3];
  const float* Wq   = (const float*)d_in[4];
  const float* Wk   = (const float*)d_in[5];
  const float* Wv   = (const float*)d_in[6];
  const float* fcw  = (const float*)d_in[7];
  const float* fcb  = (const float*)d_in[8];
  const float* g0   = (const float*)d_in[9];
  const float* b0   = (const float*)d_in[10];
  const float* g1   = (const float*)d_in[11];
  const float* b1   = (const float*)d_in[12];
  float* out = (float*)d_out;

  const size_t NLD = (size_t)4 * 2048 * 512;  // 4,194,304
  const size_t WSZ = (size_t)512 * 512;       // 262,144

  char* ws = (char*)d_ws;
  size_t off = 0;
  auto alloc = [&](size_t bytes) {
    void* p = ws + off;
    off = (off + bytes + 255) & ~(size_t)255;
    return p;
  };
  unsigned short* qbf   = (unsigned short*)alloc(NLD * 2);
  unsigned short* kbf   = (unsigned short*)alloc(NLD * 2);
  unsigned short* vbf   = (unsigned short*)alloc(NLD * 2);
  unsigned short* wqbf  = (unsigned short*)alloc(WSZ * 2);
  unsigned short* wkbf  = (unsigned short*)alloc(WSZ * 2);
  unsigned short* wvbf  = (unsigned short*)alloc(WSZ * 2);
  unsigned short* fcwbf = (unsigned short*)alloc(WSZ * 2);
  unsigned short* qhp   = (unsigned short*)alloc(NLD * 2);
  unsigned short* khp   = (unsigned short*)alloc(NLD * 2);
  unsigned short* vhTp  = (unsigned short*)alloc(NLD * 2);
  float* attn_o = (float*)alloc(NLD * 4);
  float* ln0f   = (float*)alloc(NLD * 4);
  unsigned short* ln0b = (unsigned short*)alloc(NLD * 2);
  float* fco    = (float*)alloc(NLD * 4);
  (void)ws_size; (void)n_in; (void)in_sizes; (void)out_size;

  // 1) fp32 -> bf16 conversions
  cvt_bf16_kernel<<<(int)((NLD + 255) / 256), 256, 0, stream>>>(q, qbf, (int)NLD);
  cvt_bf16_kernel<<<(int)((NLD + 255) / 256), 256, 0, stream>>>(k, kbf, (int)NLD);
  cvt_bf16_kernel<<<(int)((NLD + 255) / 256), 256, 0, stream>>>(v, vbf, (int)NLD);
  cvt_bf16_kernel<<<(int)((WSZ + 255) / 256), 256, 0, stream>>>(Wq, wqbf, (int)WSZ);
  cvt_bf16_kernel<<<(int)((WSZ + 255) / 256), 256, 0, stream>>>(Wk, wkbf, (int)WSZ);
  cvt_bf16_kernel<<<(int)((WSZ + 255) / 256), 256, 0, stream>>>(Wv, wvbf, (int)WSZ);
  cvt_bf16_kernel<<<(int)((WSZ + 255) / 256), 256, 0, stream>>>(fcw, fcwbf, (int)WSZ);

  // 2) Q/K/V projections (WMMA, 16x64 strip per wave)
  gemm_bf16_kernel<<<512, 256, 0, stream>>>(qbf, wqbf, qhp, 0);
  gemm_bf16_kernel<<<512, 256, 0, stream>>>(kbf, wkbf, khp, 0);
  gemm_bf16_kernel<<<512, 256, 0, stream>>>(vbf, wvbf, vhTp, 1);

  // 3) Flash attention (WMMA, online softmax, async double-buffered LDS)
  attn_kernel<<<dim3(32, 16), 256, 0, stream>>>(qhp, khp, vhTp, mask, attn_o);

  // 4) LN0: attn + residual(q)
  ln_kernel<<<1024, 256, 0, stream>>>(attn_o, q, nullptr, g0, b0, ln0f, ln0b);

  // 5) FC (WMMA)
  gemm_bf16_kernel<<<512, 256, 0, stream>>>(ln0b, fcwbf, fco, 2);

  // 6) LN1: fc + bias + residual(ln0) -> output
  ln_kernel<<<1024, 256, 0, stream>>>(fco, ln0f, fcb, g1, b1, out, nullptr);
}
